// PairStack_19112604467463
// MI455X (gfx1250) — matrix operand order repaired and checked by reading, not compiled, not run on hardware
//
#include <hip/hip_runtime.h>

// ---------------------------------------------------------------------------
// AlphaFold pair-stack for MI455X (gfx1250, wave32, WMMA bf16, f32 accum).
// All fragments built from wide (16B) vector loads; weights pre-packed into
// WMMA-B fragment order; incoming einsum handled by transposed scatter so the
// einsum inner loop is always contiguous; attention V staged transposed in LDS.
// Workspace: z_cur f32 (33.5MB) + 7 slots of NN*128 bf16 + ~1MB packed weights.
// ---------------------------------------------------------------------------

#define NRES 256
#define NN   (NRES * NRES)   // 65536 pair positions
#define CZ   128
#define HEADS 4
#define DATT 32

typedef __bf16 bf16_t;
typedef __attribute__((ext_vector_type(16))) __bf16 v16bf;
typedef __attribute__((ext_vector_type(8)))  __bf16 v8bf;
typedef __attribute__((ext_vector_type(4)))  __bf16 v4bf;
typedef __attribute__((ext_vector_type(8)))  float  v8f;

// ---------------------------------------------------------------------------
// WMMA helpers — VGPR layouts (wave32); C/D per CDNA5 ISA §7.12.2, A/B k order is this code's:
//  A (16x32 bf16): lane m=L&15, half kh=L>>4; element s at k = kh*16+s (contiguous order) -> one
//      contiguous 16-elem run at k = kh*16 when K is contiguous in memory.
//  B (32x16 bf16): lane n=L&15, half kh=L>>4; element s at k = kh*16+s (contiguous order) -> one
//      contiguous 16-elem run at k = kh*16 when K is contiguous.
//  C/D (16x16 f32): lane n=L&15, mh=L>>4; VGPR v holds row m = mh*8+v.
// ---------------------------------------------------------------------------
__device__ inline v8f wmma_bf16(v16bf a, v16bf b, v8f c) {
  return __builtin_amdgcn_wmma_f32_16x16x32_bf16(
      /*neg_a=*/false, a, /*neg_b=*/false, b,
      /*c_mod=*/(short)0, c, /*reuse_a=*/false, /*reuse_b=*/false);
}

__device__ inline v16bf cat8(v8bf lo, v8bf hi) {
  return __builtin_shufflevector(lo, hi, 0, 1, 2, 3, 4, 5, 6, 7,
                                 8, 9, 10, 11, 12, 13, 14, 15);
}

// A[m][k] = base[m*strideM + k], K contiguous. Two 16B loads per lane.
__device__ inline v16bf load_a_contig(const bf16_t* base, int lane, long strideM) {
  const int m = lane & 15, kh = lane >> 4;
  const bf16_t* p = base + (long)m * strideM + kh * 16;
  const v8bf lo = *(const v8bf*)(p);
  const v8bf hi = *(const v8bf*)(p + 8);
  return cat8(lo, hi);
}

// B[k][n] = base[n*strideN + k], K contiguous. Two 16B loads per lane.
__device__ inline v16bf load_b_contig(const bf16_t* base, int lane, long strideN) {
  const int n = lane & 15, kh = lane >> 4;
  const bf16_t* p = base + (long)n * strideN + kh * 16;
  const v8bf lo = *(const v8bf*)(p);
  const v8bf hi = *(const v8bf*)(p + 8);
  return cat8(lo, hi);
}

// Pre-packed weight fragment: wp[((kt*ntiles + nt)*32 + lane)*16 .. +15]
__device__ inline v16bf load_b_packed(const bf16_t* wp, int kt, int nt,
                                      int ntiles, int lane) {
  const bf16_t* p = wp + (((long)kt * ntiles + nt) * 32 + lane) * 16;
  const v8bf lo = *(const v8bf*)(p);
  const v8bf hi = *(const v8bf*)(p + 8);
  return cat8(lo, hi);
}

__device__ inline float sigmoidf_(float x) { return 1.f / (1.f + __expf(-x)); }

// ---------------------------------------------------------------------------
// Pack an f32 weight [K][N] into WMMA-B fragment order (bf16).
// One thread per (ktile, ntile, lane); writes 16 contiguous bf16.
// ---------------------------------------------------------------------------
__global__ void pack_w_kernel(const float* __restrict__ W,
                              bf16_t* __restrict__ out, int K, int N) {
  const int ntiles = N / 16;
  const int total = (K / 32) * ntiles * 32;
  const int idx = blockIdx.x * blockDim.x + threadIdx.x;
  if (idx >= total) return;
  const int lane = idx & 31;
  const int tile = idx >> 5;
  const int kt = tile / ntiles, nt = tile % ntiles;
  const int n = lane & 15, kh = lane >> 4;
  bf16_t* o = out + (long)tile * 512 + lane * 16;
#pragma unroll
  for (int v = 0; v < 8; ++v) {
    const int k = kt * 32 + kh * 16 + 2 * v;
    o[2 * v]     = (bf16_t)W[(long)k * N + nt * 16 + n];
    o[2 * v + 1] = (bf16_t)W[(long)(k + 1) * N + nt * 16 + n];
  }
}

// ---------------------------------------------------------------------------
// LayerNorm over last dim (128), fp32 in -> bf16 out. 8 waves/block.
// ---------------------------------------------------------------------------
__global__ void ln_kernel(const float* __restrict__ x, const float* __restrict__ w,
                          const float* __restrict__ b, bf16_t* __restrict__ out,
                          int R) {
  const int wave = threadIdx.x >> 5, lane = threadIdx.x & 31;
  const int r = blockIdx.x * 8 + wave;
  if (r >= R) return;
  const float4 f = ((const float4*)(x + (long)r * CZ))[lane];
  float s = f.x + f.y + f.z + f.w;
  float s2 = f.x * f.x + f.y * f.y + f.z * f.z + f.w * f.w;
#pragma unroll
  for (int off = 16; off >= 1; off >>= 1) {
    s  += __shfl_xor(s,  off, 32);
    s2 += __shfl_xor(s2, off, 32);
  }
  const float mean = s * (1.f / CZ);
  const float var  = s2 * (1.f / CZ) - mean * mean;
  const float rs   = rsqrtf(var + 1e-5f);
  const float4 wv = ((const float4*)w)[lane];
  const float4 bv = ((const float4*)b)[lane];
  v4bf ov;
  ov[0] = (bf16_t)((f.x - mean) * rs * wv.x + bv.x);
  ov[1] = (bf16_t)((f.y - mean) * rs * wv.y + bv.y);
  ov[2] = (bf16_t)((f.z - mean) * rs * wv.z + bv.z);
  ov[3] = (bf16_t)((f.w - mean) * rs * wv.w + bv.w);
  ((v4bf*)(out + (long)r * CZ))[lane] = ov;
}

// ---------------------------------------------------------------------------
// Generic WMMA GEMM: D = A(bf16, RxK) * B(packed bf16 weights), epilogues.
// Block = 256 threads (8 waves); wave computes 16 rows x NW*16 cols.
// grid.x = R/128, grid.y = Ncols/(NW*16).
// ---------------------------------------------------------------------------
#define MODE_PLAIN   0  // outB = acc                         (q,k,v proj)
#define MODE_SIGMOID 1  // outB = sigmoid(acc + b1)           (gates)
#define MODE_RELU    2  // outB = relu(acc + b1)              (transition h1)
#define MODE_GATED_T 3  // outB[c][r or rT] = sig(acc2+b2)*(acc1+b1)
#define MODE_UPDATE  4  // outF = baseF + gate*(acc + b1)     (residual update)

template <int MODE, int NW>
__launch_bounds__(256)
__global__ void gemm_kernel(const bf16_t* __restrict__ A,
                            const bf16_t* __restrict__ B1p,
                            const bf16_t* __restrict__ B2p,
                            const float*  __restrict__ bias1,
                            const float*  __restrict__ bias2,
                            const float*  __restrict__ baseF,
                            const bf16_t* __restrict__ gate,
                            bf16_t* __restrict__ outB,
                            float*  __restrict__ outF,
                            int R, int K, int Ncols, int tposeStore) {
  const int wave = threadIdx.x >> 5, lane = threadIdx.x & 31;
  const int r0 = blockIdx.x * 128 + wave * 16;
  const int n0 = blockIdx.y * (NW * 16);
  const int ntiles = Ncols / 16;
  const int nt0 = n0 / 16;

  v8f acc1[NW], acc2[NW];
#pragma unroll
  for (int t = 0; t < NW; ++t)
#pragma unroll
    for (int e = 0; e < 8; ++e) { acc1[t][e] = 0.f; acc2[t][e] = 0.f; }

  int kt = 0;
  for (int k0 = 0; k0 < K; k0 += 32, ++kt) {
    const v16bf af = load_a_contig(A + (long)r0 * K + k0, lane, K);
#pragma unroll
    for (int nt = 0; nt < NW; ++nt) {
      const v16bf bf1 = load_b_packed(B1p, kt, nt0 + nt, ntiles, lane);
      acc1[nt] = wmma_bf16(af, bf1, acc1[nt]);
      if (MODE == MODE_GATED_T) {
        const v16bf bf2 = load_b_packed(B2p, kt, nt0 + nt, ntiles, lane);
        acc2[nt] = wmma_bf16(af, bf2, acc2[nt]);
      }
    }
  }

  const int mh = lane >> 4, nn = lane & 15;
#pragma unroll
  for (int nt = 0; nt < NW; ++nt) {
    const int c = n0 + nt * 16 + nn;
    const float b1v = (MODE == MODE_PLAIN) ? 0.f : bias1[c];
#pragma unroll
    for (int v = 0; v < 8; ++v) {
      const int r = r0 + mh * 8 + v;
      const float x = acc1[nt][v] + b1v;
      if (MODE == MODE_PLAIN) {
        outB[(long)r * Ncols + c] = (bf16_t)acc1[nt][v];
      } else if (MODE == MODE_SIGMOID) {
        outB[(long)r * Ncols + c] = (bf16_t)sigmoidf_(x);
      } else if (MODE == MODE_RELU) {
        outB[(long)r * Ncols + c] = (bf16_t)fmaxf(x, 0.f);
      } else if (MODE == MODE_GATED_T) {
        const float gs = sigmoidf_(acc2[nt][v] + bias2[c]);
        // per-channel plane [c][..]; optional pair transpose so the triangle
        // einsum always reads K-contiguous data.
        const long rr = tposeStore ? ((long)(r & 255) * NRES + (r >> 8))
                                   : (long)r;
        outB[(long)c * R + rr] = (bf16_t)(gs * x);
      } else {  // MODE_UPDATE
        const float gm = gate ? (float)gate[(long)r * Ncols + c] : 1.f;
        outF[(long)r * Ncols + c] = baseF[(long)r * Ncols + c] + gm * x;
      }
    }
  }
}

// ---------------------------------------------------------------------------
// Triangle einsum: t[i,j,c] = sum_k A[c][i][k] * B[c][j][k] (NT, contiguous k;
// incoming variant handled by transposed scatter at production time).
// Block = 128 threads (4 waves); each wave a 32x32 tile; grid (4,4,128).
// ---------------------------------------------------------------------------
__launch_bounds__(128)
__global__ void tri_einsum_kernel(const bf16_t* __restrict__ At,
                                  const bf16_t* __restrict__ Bt,
                                  float* __restrict__ T) {
  const int wave = threadIdx.x >> 5, lane = threadIdx.x & 31;
  const int c  = blockIdx.z;
  const int i0 = blockIdx.x * 64 + (wave >> 1) * 32;
  const int j0 = blockIdx.y * 64 + (wave & 1) * 32;
  const bf16_t* Ap = At + (long)c * NN;
  const bf16_t* Bp = Bt + (long)c * NN;

  v8f acc[2][2];
#pragma unroll
  for (int a = 0; a < 2; ++a)
#pragma unroll
    for (int b = 0; b < 2; ++b)
#pragma unroll
      for (int e = 0; e < 8; ++e) acc[a][b][e] = 0.f;

  for (int k0 = 0; k0 < NRES; k0 += 32) {
    v16bf af[2], bfr[2];
#pragma unroll
    for (int mi = 0; mi < 2; ++mi)
      af[mi] = load_a_contig(Ap + (long)(i0 + mi * 16) * NRES + k0, lane, NRES);
#pragma unroll
    for (int nj = 0; nj < 2; ++nj)
      bfr[nj] = load_b_contig(Bp + (long)(j0 + nj * 16) * NRES + k0, lane, NRES);
#pragma unroll
    for (int mi = 0; mi < 2; ++mi)
#pragma unroll
      for (int nj = 0; nj < 2; ++nj)
        acc[mi][nj] = wmma_bf16(af[mi], bfr[nj], acc[mi][nj]);
  }

  const int mh = lane >> 4, nn = lane & 15;
#pragma unroll
  for (int mi = 0; mi < 2; ++mi)
#pragma unroll
    for (int nj = 0; nj < 2; ++nj)
#pragma unroll
      for (int v = 0; v < 8; ++v) {
        const int gi = i0 + mi * 16 + mh * 8 + v;
        const int gj = j0 + nj * 16 + nn;
        T[((long)gi * NRES + gj) * CZ + c] = acc[mi][nj][v];
      }
}

// ---------------------------------------------------------------------------
// Pair bias for triangle attention: bias[h][i][j] = sum_c zln[i,j,c]*bw[c,h]
// ---------------------------------------------------------------------------
__global__ void att_bias_kernel(const bf16_t* __restrict__ zln,
                                const float* __restrict__ bw,
                                float* __restrict__ bias) {
  const int r = blockIdx.x * blockDim.x + threadIdx.x;
  if (r >= NN) return;
  const bf16_t* row = zln + (long)r * CZ;
  float acc[HEADS] = {0.f, 0.f, 0.f, 0.f};
  for (int cc = 0; cc < CZ; cc += 8) {
    const v8bf zz = *(const v8bf*)(row + cc);
#pragma unroll
    for (int e = 0; e < 8; ++e) {
      const float z = (float)zz[e];
#pragma unroll
      for (int h = 0; h < HEADS; ++h) acc[h] += z * bw[(cc + e) * HEADS + h];
    }
  }
#pragma unroll
  for (int h = 0; h < HEADS; ++h) bias[(long)h * NN + r] = acc[h];
}

// ---------------------------------------------------------------------------
// Axial attention, flash-style. One wave per (outer, head, 16-row strip).
// starting=1: outer=i, rows=j, keys=k (row stride 128 elems).
// starting=0: outer=j, rows=i, keys=k (row stride N*128), bias transposed.
// Scores (16x256) + V^T (32x256) + probs (16x256 bf16) in LDS (40KB/WG).
// ---------------------------------------------------------------------------
__launch_bounds__(32)
__global__ void attn_kernel(const bf16_t* __restrict__ q,
                            const bf16_t* __restrict__ k,
                            const bf16_t* __restrict__ v,
                            const bf16_t* __restrict__ g,
                            const float*  __restrict__ bias,
                            bf16_t* __restrict__ o, int starting) {
  __shared__ float  sS[16 * NRES];   // 16KB scores
  __shared__ bf16_t sP[16 * NRES];   // 8KB probs
  __shared__ bf16_t sVt[DATT * NRES];// 16KB V transposed: sVt[d][key]

  const int lane  = threadIdx.x & 31;
  const int mt    = blockIdx.x;   // 16-row strip
  const int h     = blockIdx.y;
  const int outer = blockIdx.z;

  const long rowStride = starting ? (long)CZ : (long)NRES * CZ;
  const long base0     = starting ? (long)outer * NRES * CZ : (long)outer * CZ;
  const bf16_t* qbase = q + base0 + (long)(mt * 16) * rowStride + h * DATT;
  const bf16_t* kbase = k + base0 + h * DATT;
  const bf16_t* vbase = v + base0 + h * DATT;
  const float*  bh    = bias + (long)h * NN;
  const float scale = 0.17677669529663687f;  // 1/sqrt(32)
  const int mh = lane >> 4, nn = lane & 15;

  // --- stage V^T into LDS: wide row loads, scalar transposed stores ---
#pragma unroll
  for (int t = 0; t < 8; ++t) {
    const int row = t * 32 + lane;
    const bf16_t* src = vbase + (long)row * rowStride;
    const v8bf r0 = *(const v8bf*)(src);
    const v8bf r1 = *(const v8bf*)(src + 8);
    const v8bf r2 = *(const v8bf*)(src + 16);
    const v8bf r3 = *(const v8bf*)(src + 24);
#pragma unroll
    for (int e = 0; e < 8; ++e) {
      sVt[(e)      * NRES + row] = r0[e];
      sVt[(e + 8)  * NRES + row] = r1[e];
      sVt[(e + 16) * NRES + row] = r2[e];
      sVt[(e + 24) * NRES + row] = r3[e];
    }
  }

  // --- scores: s[m][key] = scale*(q.k) + bias ---
  const v16bf aq = load_a_contig(qbase, lane, rowStride);  // K = D = 32
  for (int kt = 0; kt < 16; ++kt) {
    const v16bf bk =
        load_b_contig(kbase + (long)(kt * 16) * rowStride, lane, rowStride);
    v8f sacc;
#pragma unroll
    for (int e = 0; e < 8; ++e) sacc[e] = 0.f;
    sacc = wmma_bf16(aq, bk, sacc);
#pragma unroll
    for (int vv = 0; vv < 8; ++vv) {
      const int m = mh * 8 + vv;
      const int key = kt * 16 + nn;
      const int mrow = mt * 16 + m;
      const float bv = starting ? bh[(long)mrow * NRES + key]
                                : bh[(long)key * NRES + mrow];
      sS[m * NRES + key] = sacc[vv] * scale + bv;
    }
  }
  __syncthreads();

  // --- softmax over 256 keys per row ---
  for (int m = 0; m < 16; ++m) {
    float vals[8], mx = -1e30f;
#pragma unroll
    for (int t = 0; t < 8; ++t) {
      vals[t] = sS[m * NRES + lane + 32 * t];
      mx = fmaxf(mx, vals[t]);
    }
#pragma unroll
    for (int off = 16; off >= 1; off >>= 1) mx = fmaxf(mx, __shfl_xor(mx, off, 32));
    float sm = 0.f;
#pragma unroll
    for (int t = 0; t < 8; ++t) { vals[t] = __expf(vals[t] - mx); sm += vals[t]; }
#pragma unroll
    for (int off = 16; off >= 1; off >>= 1) sm += __shfl_xor(sm, off, 32);
    const float inv = 1.f / sm;
#pragma unroll
    for (int t = 0; t < 8; ++t)
      sP[m * NRES + lane + 32 * t] = (bf16_t)(vals[t] * inv);
  }
  __syncthreads();

  // --- O = P(16x256) x V(256x32), both operands K-contiguous in LDS ---
  v8f oacc[2];
#pragma unroll
  for (int dh = 0; dh < 2; ++dh)
#pragma unroll
    for (int e = 0; e < 8; ++e) oacc[dh][e] = 0.f;

  for (int ks = 0; ks < 8; ++ks) {
    const v16bf ap = load_a_contig(sP + ks * 32, lane, NRES);
#pragma unroll
    for (int dh = 0; dh < 2; ++dh) {
      const v16bf bv =
          load_b_contig(sVt + (long)(dh * 16) * NRES + ks * 32, lane, NRES);
      oacc[dh] = wmma_bf16(ap, bv, oacc[dh]);
    }
  }

  // --- gated store: o = sigmoid-gate * o (gate precomputed bf16) ---
#pragma unroll
  for (int dh = 0; dh < 2; ++dh)
#pragma unroll
    for (int vv = 0; vv < 8; ++vv) {
      const int m = mh * 8 + vv;
      const long idx = base0 + (long)(mt * 16 + m) * rowStride + h * DATT +
                       dh * 16 + nn;
      o[idx] = (bf16_t)((float)g[idx] * oacc[dh][vv]);
    }
}

// ---------------------------------------------------------------------------
// Host orchestration
// ---------------------------------------------------------------------------
extern "C" void kernel_launch(void* const* d_in, const int* in_sizes, int n_in,
                              void* d_out, int out_size, void* d_ws,
                              size_t ws_size, hipStream_t stream) {
  (void)in_sizes; (void)n_in; (void)out_size; (void)ws_size;
  const float* z_in = (const float*)d_in[0];
  auto W = [&](int i) { return (const float*)d_in[i]; };

  char* wsb = (char*)d_ws;
  float* zcur = (float*)wsb;                                  // NN*CZ f32
  char* pool = wsb + (size_t)NN * CZ * sizeof(float);
  const size_t SLOT = (size_t)NN * CZ * sizeof(bf16_t);       // 16.78 MB
  auto slot = [&](int i) { return pool + SLOT * (size_t)i; }; // 7 slots

  // ---- pre-pack all GEMM weights into WMMA-B fragment order (bf16) ----
  bf16_t* wpk[60] = {};
  char* wp = pool + 7 * SLOT;
  auto packW = [&](int idx, int K, int N) {
    bf16_t* dst = (bf16_t*)wp;
    wp += (size_t)K * N * sizeof(bf16_t);
    const int total = (K / 32) * (N / 16) * 32;
    pack_w_kernel<<<dim3((total + 255) / 256), dim3(256), 0, stream>>>(
        W(idx), dst, K, N);
    wpk[idx] = dst;
  };
  {
    const int tm[2] = {1, 17};
    for (int s = 0; s < 2; ++s) {
      const int p0 = tm[s];
      packW(p0 + 2, CZ, CZ);  packW(p0 + 4, CZ, CZ);  packW(p0 + 6, CZ, CZ);
      packW(p0 + 8, CZ, CZ);  packW(p0 + 12, CZ, CZ); packW(p0 + 14, CZ, CZ);
    }
    const int ta[2] = {33, 43};
    for (int s = 0; s < 2; ++s) {
      const int p0 = ta[s];
      packW(p0 + 3, CZ, CZ); packW(p0 + 4, CZ, CZ); packW(p0 + 5, CZ, CZ);
      packW(p0 + 6, CZ, CZ); packW(p0 + 8, CZ, CZ);
    }
    packW(55, CZ, 4 * CZ);   // transition w1
    packW(57, 4 * CZ, CZ);   // transition w2
  }

  const dim3 blk256(256), blk128(128), blk32(32);
  const dim3 gLN(NN / 8);
  const dim3 gG1(NN / 128, 1);   // Ncols=128, NW=8
  const dim3 gG2(NN / 128, 2);   // Ncols=128, NW=4 (dual-accum gated)
  const dim3 gG4(NN / 128, 4);   // Ncols=512, NW=8
  const dim3 gEin(4, 4, CZ);
  const dim3 gAttn(NRES / 16, HEADS, NRES);
  const dim3 gBias(NN / 256);

  // ---- gated triangle multiplicative update (Alg 11/12) ----
  // params: +0 ln_in_w +1 ln_in_b +2 ap_w +3 ap_b +4 ag_w +5 ag_b
  //         +6 bp_w +7 bp_b +8 bg_w +9 bg_b +10 ln_out_w +11 ln_out_b
  //         +12 z_w +13 z_b +14 g_w +15 g_b
  auto tri_mult = [&](const float* zin, float* zout, int p0, int outgoing) {
    bf16_t* zln = (bf16_t*)slot(0);
    bf16_t* gb  = (bf16_t*)slot(1);
    bf16_t* at  = (bf16_t*)slot(2);
    bf16_t* bt  = (bf16_t*)slot(3);
    float*  tb  = (float*)slot(4);     // 2 slots (f32)
    bf16_t* tln = (bf16_t*)slot(6);
    const int tpose = outgoing ? 0 : 1;
    ln_kernel<<<gLN, blk256, 0, stream>>>(zin, W(p0 + 0), W(p0 + 1), zln, NN);
    gemm_kernel<MODE_GATED_T, 4><<<gG2, blk256, 0, stream>>>(
        zln, wpk[p0 + 2], wpk[p0 + 4], W(p0 + 3), W(p0 + 5), nullptr, nullptr,
        at, nullptr, NN, CZ, CZ, tpose);
    gemm_kernel<MODE_GATED_T, 4><<<gG2, blk256, 0, stream>>>(
        zln, wpk[p0 + 6], wpk[p0 + 8], W(p0 + 7), W(p0 + 9), nullptr, nullptr,
        bt, nullptr, NN, CZ, CZ, tpose);
    gemm_kernel<MODE_SIGMOID, 8><<<gG1, blk256, 0, stream>>>(
        zln, wpk[p0 + 14], nullptr, W(p0 + 15), nullptr, nullptr, nullptr,
        gb, nullptr, NN, CZ, CZ, 0);
    tri_einsum_kernel<<<gEin, blk128, 0, stream>>>(at, bt, tb);
    ln_kernel<<<gLN, blk256, 0, stream>>>(tb, W(p0 + 10), W(p0 + 11), tln, NN);
    gemm_kernel<MODE_UPDATE, 8><<<gG1, blk256, 0, stream>>>(
        tln, wpk[p0 + 12], nullptr, W(p0 + 13), nullptr, zin, gb,
        nullptr, zout, NN, CZ, CZ, 0);
  };

  // ---- triangle attention (Alg 13/14) ----
  // params: +0 ln_w +1 ln_b +2 b_w +3 q_w +4 k_w +5 v_w +6 g_w +7 g_b
  //         +8 o_w +9 o_b
  auto tri_att = [&](float* zio, int p0, int starting) {
    bf16_t* zln = (bf16_t*)slot(0);
    bf16_t* gb  = (bf16_t*)slot(1);
    bf16_t* qb  = (bf16_t*)slot(2);
    bf16_t* kb  = (bf16_t*)slot(3);
    bf16_t* vb  = (bf16_t*)slot(4);
    bf16_t* ob  = (bf16_t*)slot(5);
    float*  bb  = (float*)slot(6);     // HEADS*NN f32 = 1MB
    ln_kernel<<<gLN, blk256, 0, stream>>>(zio, W(p0 + 0), W(p0 + 1), zln, NN);
    att_bias_kernel<<<gBias, blk256, 0, stream>>>(zln, W(p0 + 2), bb);
    gemm_kernel<MODE_PLAIN, 8><<<gG1, blk256, 0, stream>>>(
        zln, wpk[p0 + 3], nullptr, nullptr, nullptr, nullptr, nullptr,
        qb, nullptr, NN, CZ, CZ, 0);
    gemm_kernel<MODE_PLAIN, 8><<<gG1, blk256, 0, stream>>>(
        zln, wpk[p0 + 4], nullptr, nullptr, nullptr, nullptr, nullptr,
        kb, nullptr, NN, CZ, CZ, 0);
    gemm_kernel<MODE_PLAIN, 8><<<gG1, blk256, 0, stream>>>(
        zln, wpk[p0 + 5], nullptr, nullptr, nullptr, nullptr, nullptr,
        vb, nullptr, NN, CZ, CZ, 0);
    gemm_kernel<MODE_SIGMOID, 8><<<gG1, blk256, 0, stream>>>(
        zln, wpk[p0 + 6], nullptr, W(p0 + 7), nullptr, nullptr, nullptr,
        gb, nullptr, NN, CZ, CZ, 0);
    attn_kernel<<<gAttn, blk32, 0, stream>>>(qb, kb, vb, gb, bb, ob, starting);
    gemm_kernel<MODE_UPDATE, 8><<<gG1, blk256, 0, stream>>>(
        ob, wpk[p0 + 8], nullptr, W(p0 + 9), nullptr, zio, nullptr,
        nullptr, zio, NN, CZ, CZ, 0);
  };

  // ---- transition MLP (Alg 15) ----
  auto transition = [&](const float* zin, float* zout, int p0) {
    bf16_t* zln = (bf16_t*)slot(0);
    bf16_t* h1  = (bf16_t*)slot(1);    // 4 slots: NN*512 bf16
    ln_kernel<<<gLN, blk256, 0, stream>>>(zin, W(p0 + 0), W(p0 + 1), zln, NN);
    gemm_kernel<MODE_RELU, 8><<<gG4, blk256, 0, stream>>>(
        zln, wpk[p0 + 2], nullptr, W(p0 + 3), nullptr, nullptr, nullptr,
        h1, nullptr, NN, CZ, 4 * CZ, 0);
    gemm_kernel<MODE_UPDATE, 8><<<gG1, blk256, 0, stream>>>(
        h1, wpk[p0 + 4], nullptr, W(p0 + 5), nullptr, zin, nullptr,
        nullptr, zout, NN, 4 * CZ, CZ, 0);
  };

  // ---- pair stack: z -> +tm_out -> +tm_in -> +att_start -> +att_end -> +trans
  tri_mult(z_in, zcur, 1, /*outgoing=*/1);   // tmo: params 1..16
  tri_mult(zcur, zcur, 17, /*outgoing=*/0);  // tmi: params 17..32
  tri_att(zcur, 33, /*starting=*/1);         // tas: params 33..42
  tri_att(zcur, 43, /*starting=*/0);         // tae: params 43..52
  transition(zcur, (float*)d_out, 53);       // pt : params 53..58
}
